// myEncoder_ver_3_20495583936825
// MI455X (gfx1250) — matrix operand order repaired and checked by reading, not compiled, not run on hardware
//
#include <hip/hip_runtime.h>
#include <hip/hip_bf16.h>
#include <stdint.h>

// Problem constants (match reference)
#define B_   64
#define T_   1024
#define E_   512
#define H_   512
#define NS_  2
#define GH_  2048          // 4*H
#define NWG  64            // persistent workgroups (one gate-group of 8 hidden units each)
#define NTHR 256           // 8 waves of 32

typedef __attribute__((ext_vector_type(16))) __bf16        v16bf;
typedef __attribute__((ext_vector_type(8)))  float         v8f;
typedef __attribute__((ext_vector_type(8)))  unsigned short v8us;

struct V8x2 { v8us lo, hi; };
union ABf { V8x2 u; v16bf bf; };

// fp32 -> bf16 round-to-nearest-even
__device__ __forceinline__ unsigned short f2bf(float f) {
  unsigned u = __float_as_uint(f);
  unsigned r = u + 0x7FFFu + ((u >> 16) & 1u);
  return (unsigned short)(r >> 16);
}

__device__ __forceinline__ float sigm(float x) { return 1.0f / (1.0f + __expf(-x)); }

// Packed-weight fragment index: [s][mat][ntile(128)][kc(16)][lane(32)][16 bf16]
__device__ __forceinline__ size_t widx(int s, int mat, int ntile, int kc, int lane) {
  return ((((size_t)((s * 2 + mat) * 128 + ntile)) * 16 + kc) * 32 + lane) * 16;
}

#define WMAT_STRIDE 1048576u   // widx stride between mat=0 and mat=1 (elements)
#define WKC_STRIDE  512u       // widx stride between consecutive kc (elements)

// -------------------------------------------------------------------------
// Prep: pack weights fp32->bf16 in WMMA B-fragment order with gate-row
// permutation (i,f,g,o interleaved per 8 hidden units), fold biases,
// zero h/c state + bf16 h buffers + barrier counters. Runs every launch.
// -------------------------------------------------------------------------
__global__ void prep_kernel(const float* __restrict__ W_ih, const float* __restrict__ W_hh,
                            const float* __restrict__ b_ih, const float* __restrict__ b_hh,
                            unsigned short* __restrict__ Wpack, float* __restrict__ biaspack,
                            unsigned short* __restrict__ hb16,
                            float* __restrict__ out_h, float* __restrict__ out_c,
                            unsigned* __restrict__ bar) {
  size_t tid = (size_t)blockIdx.x * blockDim.x + threadIdx.x;
  const size_t NW = (size_t)NS_ * 2 * 128 * 16 * 32 * 16;   // 4,194,304 packed elems
  if (tid < NW) {
    size_t f = tid;
    int v    = (int)(f & 15);  f >>= 4;
    int lane = (int)(f & 31);  f >>= 5;
    int kc   = (int)(f & 15);  f >>= 4;
    int nt   = (int)(f & 127); f >>= 7;
    int mat  = (int)(f & 1);
    int s    = (int)(f >> 1);
    int c    = nt * 16 + (lane & 15);                 // permuted gate column
    int k    = kc * 32 + ((lane >> 4) & 1) * 16 + v;  // K position
    int grp  = c >> 5, rem = c & 31;
    int gate = rem >> 3, j7 = rem & 7;
    int row  = gate * H_ + grp * 8 + j7;              // original weight row
    const float* W = mat ? W_hh : W_ih;
    Wpack[tid] = f2bf(W[((size_t)s * GH_ + row) * E_ + k]);
  }
  if (tid < (size_t)(NS_ * GH_)) {
    int s = (int)(tid >> 11), c = (int)(tid & (GH_ - 1));
    int grp = c >> 5, rem = c & 31;
    int gate = rem >> 3, j7 = rem & 7;
    int row = gate * H_ + grp * 8 + j7;
    biaspack[tid] = b_ih[s * GH_ + row] + b_hh[s * GH_ + row];
  }
  if (tid < (size_t)NS_ * 2 * B_ * H_) hb16[tid] = 0;      // both t-parities
  if (tid < (size_t)NS_ * B_ * H_) { out_h[tid] = 0.0f; out_c[tid] = 0.0f; }
  if (tid == 0) { bar[0] = 0; bar[1] = 0; }
}

// -------------------------------------------------------------------------
// Persistent recurrent kernel: one grid barrier per (t, stack) phase.
// Each WG owns 32 gate columns = 8 hidden units x {i,f,g,o}; 8 waves each
// compute one 16x16 gate tile via v_wmma_f32_16x16x32_bf16. All 8 weight
// (B) fragments of a K-chunk are prefetched into live registers behind a
// sched_barrier so their 16 global_load_b128s overlap a single L2 latency.
// -------------------------------------------------------------------------
__global__ void __launch_bounds__(NTHR) lstm_kernel(
    const float* __restrict__ seq, const int* __restrict__ seq_len,
    const unsigned short* __restrict__ Wpack, const float* __restrict__ biaspack,
    unsigned short* __restrict__ hb16, float* __restrict__ inpB,
    float* __restrict__ out_hid, float* __restrict__ out_h, float* __restrict__ out_c,
    unsigned* __restrict__ bar)
{
  __shared__ unsigned short xsh[B_ * 128];   // bf16 x-chunk  (16 KB)
  __shared__ unsigned short hsh[B_ * 128];   // bf16 h-chunk  (16 KB)
  __shared__ float          gsh[B_ * 32];    // gate tile     (8 KB)

  const int tid     = threadIdx.x;
  const int wv      = tid >> 5, ln = tid & 31;
  const int mt      = wv >> 1, ntl = wv & 1;
  const int ntile_g = blockIdx.x * 2 + ntl;
  const int arow    = mt * 16 + (ln & 15);
  const int kb      = (ln >= 16) ? 16 : 0;         // A-fragment K sub-offset
  const int myc     = ntile_g * 16 + (ln & 15);    // my gate column

  for (int t = 0; t < T_; ++t) {
    for (int s = 0; s < NS_; ++s) {
      v8f acc = {};
      const float* inp32   = (s == 0) ? (seq + (size_t)t * E_) : inpB;
      const size_t rstride = (s == 0) ? (size_t)T_ * E_ : (size_t)E_;
      const unsigned short* hsrc = hb16 + ((size_t)(s * 2 + (t & 1))) * B_ * H_;

      for (int ch = 0; ch < 4; ++ch) {             // K chunks of 128
        __syncthreads();
        // Cooperative staging: x fp32->bf16 convert, h bf16 copy
        #pragma unroll
        for (int i = 0; i < (B_ * 128) / (4 * NTHR); ++i) {
          int fl  = (tid + i * NTHR) * 4;
          int row = fl >> 7, col = fl & 127;
          const float4 xf = *(const float4*)(inp32 + (size_t)row * rstride + ch * 128 + col);
          ushort4 xb;
          xb.x = f2bf(xf.x); xb.y = f2bf(xf.y); xb.z = f2bf(xf.z); xb.w = f2bf(xf.w);
          *(ushort4*)(&xsh[row * 128 + col]) = xb;
          *(ushort4*)(&hsh[row * 128 + col]) =
              *(const ushort4*)(hsrc + (size_t)row * H_ + ch * 128 + col);
        }
        __syncthreads();

        // 8 steps: st = kc*2 + mat (mat 0 = x*W_ih, 1 = h*W_hh)
        const unsigned short* abase_x = &xsh[arow * 128 + kb];
        const unsigned short* abase_h = &hsh[arow * 128 + kb];
        const unsigned short* wbase   = Wpack + widx(s, 0, ntile_g, ch * 4, ln);

        // Prefetch ALL 8 B fragments (16x global_load_b128 in flight at once).
        ABf bfr[8];
        #pragma unroll
        for (int st = 0; st < 8; ++st) {
          const int mat_ = st & 1, kc_ = st >> 1;
          const unsigned short* bp_ =
              wbase + (size_t)mat_ * WMAT_STRIDE + (size_t)kc_ * WKC_STRIDE;
          bfr[st].u.lo = *(const v8us*)bp_;
          bfr[st].u.hi = *(const v8us*)(bp_ + 8);
        }
        // Hard fence: no WMMA/ds may be hoisted above, no B-load sunk below.
        __builtin_amdgcn_sched_barrier(0);

        #pragma unroll
        for (int st = 0; st < 8; ++st) {
          const int mat_ = st & 1, kc_ = st >> 1;
          const unsigned short* ap_ = (mat_ ? abase_h : abase_x) + kc_ * 32;
          ABf a;
          a.u.lo = *(const v8us*)ap_;
          a.u.hi = *(const v8us*)(ap_ + 8);
          acc = __builtin_amdgcn_wmma_f32_16x16x32_bf16(
                    false, a.bf, false, bfr[st].bf, (short)0, acc, false, false);
        }
      }

      // Bias + deposit gate tile to LDS (C-layout: VGPR r -> rows r / 8+r)
      {
        const float bv    = biaspack[s * GH_ + myc];
        const int   crow0 = mt * 16 + ((ln >= 16) ? 8 : 0);
        const int   ccol  = ntl * 16 + (ln & 15);
        #pragma unroll
        for (int r = 0; r < 8; ++r)
          gsh[(crow0 + r) * 32 + ccol] = acc[r] + bv;
      }
      __syncthreads();

      // WG-local LSTM cell update: 64 batch x 8 hidden units (2 per thread)
      #pragma unroll
      for (int kk = 0; kk < 2; ++kk) {
        int cidx = tid + kk * NTHR;
        int b = cidx >> 3, jl = cidx & 7;
        float gi = gsh[b * 32 + jl];
        float gf = gsh[b * 32 + 8 + jl];
        float gg = gsh[b * 32 + 16 + jl];
        float go = gsh[b * 32 + 24 + jl];
        int j = blockIdx.x * 8 + jl;
        size_t sij = (size_t)s * B_ * H_ + (size_t)b * H_ + j;
        float c_old = out_c[sij];
        float h_old = out_h[sij];
        float cs = sigm(gf) * c_old + sigm(gi) * tanhf(gg);
        float hs = sigm(go) * tanhf(cs);
        bool  m  = t < seq_len[b];
        float cv = m ? cs : c_old;
        float hv = m ? hs : h_old;
        out_c[sij] = cv;
        out_h[sij] = hv;
        hb16[((size_t)(s * 2 + ((t + 1) & 1))) * B_ * H_ + (size_t)b * H_ + j] = f2bf(hv);
        float ip = (s == 0) ? seq[(size_t)b * T_ * E_ + (size_t)t * E_ + j]
                            : inpB[b * E_ + j];
        float ni = hs + ip;                         // residual (unmasked, per reference)
        if (s == 0) inpB[b * E_ + j] = ni;
        else        out_hid[(size_t)b * T_ * E_ + (size_t)t * E_ + j] = m ? ni : 0.0f;
      }

      // ---- grid-wide barrier (L2 atomics, sense via generation counter) ----
      __syncthreads();
      if (tid == 0) {
        __threadfence();
        volatile unsigned* gen = bar + 1;
        unsigned g = *gen;
        if (atomicAdd(bar, 1u) == NWG - 1) {
          atomicExch(bar, 0u);
          __threadfence();
          atomicAdd((unsigned*)(bar + 1), 1u);
        } else {
          while (*gen == g) { __builtin_amdgcn_s_sleep(1); }
        }
        __threadfence();
      }
      __syncthreads();
    }
  }
}

// -------------------------------------------------------------------------
extern "C" void kernel_launch(void* const* d_in, const int* in_sizes, int n_in,
                              void* d_out, int out_size, void* d_ws, size_t ws_size,
                              hipStream_t stream) {
  const float* seq  = (const float*)d_in[0];
  const int*   slen = (const int*)d_in[1];
  const float* W_ih = (const float*)d_in[2];
  const float* W_hh = (const float*)d_in[3];
  const float* b_ih = (const float*)d_in[4];
  const float* b_hh = (const float*)d_in[5];

  float* out     = (float*)d_out;
  float* out_hid = out;                                   // [B,T,E]
  float* out_h   = out + (size_t)B_ * T_ * E_;            // [NS,B,H]
  float* out_c   = out_h + (size_t)NS_ * B_ * H_;         // [NS,B,H]

  uint8_t* ws = (uint8_t*)d_ws;
  unsigned*       bar      = (unsigned*)ws;                               // 8 B (+pad)
  unsigned short* Wpack    = (unsigned short*)(ws + 256);                 // 8 MB
  float*          biaspack = (float*)(ws + 256 + 8388608);                // 16 KB
  unsigned short* hb16     = (unsigned short*)(ws + 256 + 8388608 + 16384);          // 256 KB
  float*          inpB     = (float*)(ws + 256 + 8388608 + 16384 + 262144);          // 128 KB

  prep_kernel<<<16384, 256, 0, stream>>>(W_ih, W_hh, b_ih, b_hh,
                                         Wpack, biaspack, hb16, out_h, out_c, bar);
  lstm_kernel<<<NWG, NTHR, 0, stream>>>(seq, slen, Wpack, biaspack, hb16, inpB,
                                        out_hid, out_h, out_c, bar);
}
